// LatentHypergraphSDE_4088808866146
// MI455X (gfx1250) — matrix operand order repaired and checked by reading, not compiled, not run on hardware
//
#include <hip/hip_runtime.h>
#include <hip/hip_bf16.h>
#include <stdint.h>

// ---------------------------------------------------------------------------
// LatentHypergraphSDE on MI455X (gfx1250, wave32, WMMA)
//   encode -> 20x (bf16 WMMA GEMM, scatter n->e, scatter e->n, SDE update)
//          -> decode
// GEMM A/B operands are kept as packed bf16 so each WMMA operand is two
// global_load_b128's with no conversion ALU. f32 accumulate throughout.
// Scatter adds use native global_atomic_add_f32. Working set is L2-resident.
// ---------------------------------------------------------------------------

typedef __attribute__((ext_vector_type(16))) __bf16    v16bf;
typedef __attribute__((ext_vector_type(8)))  float     v8f;
typedef __attribute__((ext_vector_type(8)))  uint32_t  v8u;

#define LATD 64
#define OBSD 64
#define EDGES 12500
#define SDE_STEPS 20
#define SDE_DT 0.05f
#define SDE_SQRT_DT 0.22360679774997896f
#define WPB 4   // waves per block for GEMM kernels (4 * 32 = 128 threads)

// ----------------------------- helpers -------------------------------------

static __device__ __forceinline__ uint32_t f2bf_bits(float f) {
  uint32_t u = __builtin_bit_cast(uint32_t, f);
  return (u + 0x7FFFu + ((u >> 16) & 1u)) >> 16;   // round-to-nearest-even
}
static __device__ __forceinline__ __bf16 f2bf(float f) {
  return __builtin_bit_cast(__bf16, (uint16_t)f2bf_bits(f));
}
static __device__ __forceinline__ uint32_t pack2bf(float lo, float hi) {
  return f2bf_bits(lo) | (f2bf_bits(hi) << 16);
}

static __device__ __forceinline__ void atomicAddF(float* p, float v) {
  unsafeAtomicAdd(p, v);   // emits global_atomic_add_f32
}

// A-matrix (16x32 bf16, MxK) per CDNA5 ISA 7.12.2. With packed bf16 rows the
// lane's 8 K-pairs are dwords [d0..d0+3] and [d0+8..d0+11]: two b128 loads.
static __device__ __forceinline__ v16bf load_A_packed(const uint32_t* __restrict__ bf,
                                                      int row, int kbase, int lane) {
  const uint32_t* rp = bf + (size_t)row * 32;               // 64 bf16 = 32 dwords
  const int d0 = (kbase + ((lane >> 4) << 3)) >> 1;          // {0,4,16,20}
  uint4 q0 = *(const uint4*)(rp + d0);
  uint4 q1 = *(const uint4*)(rp + d0 + 8);
  v8u t = {q0.x, q0.y, q0.z, q0.w, q1.x, q1.y, q1.z, q1.w};
  return __builtin_bit_cast(v16bf, t);
}

// B-matrix (32x16 bf16, KxN) for y = x @ W^T: lane N = L&15, K pairs of row
// W[n] are dwords [d0..d0+3] and [d0+8..d0+11] (same as A): two b128 loads.
static __device__ __forceinline__ v16bf load_B_packed(const uint32_t* __restrict__ Wbf,
                                                      int ntile, int kbase, int lane) {
  const uint32_t* rp = Wbf + (size_t)((ntile << 4) + (lane & 15)) * 32
                          + ((kbase + ((lane >> 4) << 3)) >> 1);   // {0,4,16,20}
  uint4 q0 = *(const uint4*)(rp);
  uint4 q1 = *(const uint4*)(rp + 8);
  v8u t = {q0.x, q0.y, q0.z, q0.w, q1.x, q1.y, q1.z, q1.w};
  return __builtin_bit_cast(v16bf, t);
}

// A from an f32 LDS tile (stride 65 floats), converting on the fly (cold path)
static __device__ __forceinline__ v16bf load_A_lds_f32(const float* base,
                                                       int kbase, int lane) {
  const float* rp = base + (lane & 15) * 65;
  const int khalf = (lane >> 4) << 3;
  v16bf a;
#pragma unroll
  for (int e = 0; e < 16; ++e) {
    int p = e >> 1;
    int k = kbase + ((p >> 2) << 4) + khalf + ((p & 3) << 1) + (e & 1);
    a[e] = f2bf(rp[k]);
  }
  return a;
}

#define WMMA_BF16(a, b, c) \
  __builtin_amdgcn_wmma_f32_16x16x32_bf16(false, (a), false, (b), (short)0, (c), false, false)

// counter-based Gaussian (deterministic per call; stands in for jax threefry)
static __device__ __forceinline__ uint32_t hash3(uint32_t x, uint32_t y, uint32_t z) {
  uint32_t h = x * 0x9E3779B1u ^ y * 0x85EBCA77u ^ z * 0xC2B2AE3Du;
  h ^= h >> 16; h *= 0x7FEB352Du; h ^= h >> 15; h *= 0x846CA68Bu; h ^= h >> 16;
  return h;
}
static __device__ __forceinline__ float gauss1(uint32_t seed, uint32_t step, uint32_t idx) {
  uint32_t h1 = hash3(seed ^ 0xA511E9B3u, step, idx);
  uint32_t h2 = hash3(seed ^ 0x63D83595u, step + 0x9E3779B9u, idx);
  float u1 = (float)(h1 >> 8) * (1.0f / 16777216.0f) + (0.5f / 16777216.0f); // (0,1]
  float u2 = (float)(h2 >> 8) * (1.0f / 16777216.0f);
  return sqrtf(-2.0f * __logf(u1)) * __cosf(6.28318530717958647f * u2);
}

// ----------------------------- kernels -------------------------------------

// f32 -> packed bf16 (pairs); used for x, weights, and y after encode
__global__ void hsde_cvt_bf16(const float* __restrict__ src,
                              uint32_t* __restrict__ dst, int npairs) {
  int i = blockIdx.x * blockDim.x + threadIdx.x;
  if (i < npairs) {
    float2 v = ((const float2*)src)[i];
    dst[i] = pack2bf(v.x, v.y);
  }
}

// m[N,64] = y[N,64] @ W^T   (bf16 inputs, f32 out) — one 16-row tile per wave
__global__ void hsde_conv_mm(const uint32_t* __restrict__ ybf,
                             const uint32_t* __restrict__ Wbf,
                             float* __restrict__ m, int Nn) {
  const int lane = threadIdx.x & 31;
  const int wid  = blockIdx.x * (blockDim.x >> 5) + (threadIdx.x >> 5);
  const int rowbase = wid << 4;
  if (rowbase >= Nn) return;

  int arow = rowbase + (lane & 15);
  if (arow >= Nn) arow = Nn - 1;
  v16bf a0 = load_A_packed(ybf, arow, 0,  lane);
  v16bf a1 = load_A_packed(ybf, arow, 32, lane);

  const bool full = (rowbase + 16 <= Nn);
  float* pr = m + (size_t)(rowbase + ((lane >> 4) << 3)) * LATD + (lane & 15);

#pragma unroll
  for (int nt = 0; nt < 4; ++nt) {
    v8f c = {};
    c = WMMA_BF16(a0, load_B_packed(Wbf, nt, 0,  lane), c);
    c = WMMA_BF16(a1, load_B_packed(Wbf, nt, 32, lane), c);
    if (full) {
#pragma unroll
      for (int r = 0; r < 8; ++r) pr[nt * 16 + r * LATD] = c[r];   // imm offsets
    } else {
      const int col = (nt << 4) + (lane & 15);
#pragma unroll
      for (int r = 0; r < 8; ++r) {
        int M = rowbase + r + ((lane >> 4) << 3);
        if (M < Nn) m[(size_t)M * LATD + col] = c[r];
      }
    }
  }
}

// out[N,64] = relu(Xbf @ W1^T + b1) @ W2^T + b2   (encoder / decoder)
__global__ void hsde_mlp2(const uint32_t* __restrict__ Xbf,
                          const uint32_t* __restrict__ W1bf, const float* __restrict__ b1,
                          const uint32_t* __restrict__ W2bf, const float* __restrict__ b2,
                          float* __restrict__ out, int Nn) {
  __shared__ float hb[WPB][16][65];   // stride-65 padding: conflict-free LDS
  const int lane = threadIdx.x & 31;
  const int wv   = threadIdx.x >> 5;
  const int wid  = blockIdx.x * (blockDim.x >> 5) + wv;
  const int rowbase = wid << 4;
  if (rowbase >= Nn) return;

  // layer 1: hidden = relu(X @ W1^T + b1), parked in LDS
  {
    int arow = rowbase + (lane & 15);
    if (arow >= Nn) arow = Nn - 1;
    v16bf a0 = load_A_packed(Xbf, arow, 0,  lane);
    v16bf a1 = load_A_packed(Xbf, arow, 32, lane);
#pragma unroll
    for (int nt = 0; nt < 4; ++nt) {
      v8f c = {};
      c = WMMA_BF16(a0, load_B_packed(W1bf, nt, 0,  lane), c);
      c = WMMA_BF16(a1, load_B_packed(W1bf, nt, 32, lane), c);
      const int col = (nt << 4) + (lane & 15);
      const float bias = b1[col];
#pragma unroll
      for (int r = 0; r < 8; ++r) {
        int Ml = r + ((lane >> 4) << 3);
        hb[wv][Ml][col] = fmaxf(c[r] + bias, 0.0f);
      }
    }
  }
  // layer 2: out = hidden @ W2^T + b2
  {
    v16bf a0 = load_A_lds_f32(&hb[wv][0][0], 0,  lane);
    v16bf a1 = load_A_lds_f32(&hb[wv][0][0], 32, lane);
    const bool full = (rowbase + 16 <= Nn);
    float* pr = out + (size_t)(rowbase + ((lane >> 4) << 3)) * LATD + (lane & 15);
#pragma unroll
    for (int nt = 0; nt < 4; ++nt) {
      v8f c = {};
      c = WMMA_BF16(a0, load_B_packed(W2bf, nt, 0,  lane), c);
      c = WMMA_BF16(a1, load_B_packed(W2bf, nt, 32, lane), c);
      const int col = (nt << 4) + (lane & 15);
      const float bias = b2[col];
      if (full) {
#pragma unroll
        for (int r = 0; r < 8; ++r) pr[nt * 16 + r * LATD] = c[r] + bias;
      } else {
#pragma unroll
        for (int r = 0; r < 8; ++r) {
          int M = rowbase + r + ((lane >> 4) << 3);
          if (M < Nn) out[(size_t)M * LATD + col] = c[r] + bias;
        }
      }
    }
  }
}

// degree counting + inversion
__global__ void hsde_count(const int* __restrict__ node_idx, const int* __restrict__ edge_idx,
                           float* __restrict__ node_cnt, float* __restrict__ edge_cnt, int nnz) {
  int i = blockIdx.x * blockDim.x + threadIdx.x;
  if (i < nnz) {
    atomicAddF(&edge_cnt[edge_idx[i]], 1.0f);
    atomicAddF(&node_cnt[node_idx[i]], 1.0f);
  }
}
__global__ void hsde_invert(float* __restrict__ v, int n) {
  int i = blockIdx.x * blockDim.x + threadIdx.x;
  if (i < n) v[i] = 1.0f / fmaxf(v[i], 1.0f);
}

// dst[scatter_idx[i]] += src[gather_idx[i]] * (scale ? scale[gather_idx[i]] : 1)
__global__ void hsde_scatter(const float* __restrict__ src,
                             const int* __restrict__ gather_idx,
                             const int* __restrict__ scatter_idx,
                             const float* __restrict__ scale,
                             float* __restrict__ dst, int nnz) {
  int tid = blockIdx.x * blockDim.x + threadIdx.x;
  int i = tid >> 4;
  if (i >= nnz) return;
  int cg = (tid & 15) << 2;
  int g = gather_idx[i];
  int s = scatter_idx[i];
  float sc = scale ? scale[g] : 1.0f;
  const float4 v = *(const float4*)(src + (size_t)g * LATD + cg);
  float* d = dst + (size_t)s * LATD + cg;
  atomicAddF(d + 0, v.x * sc);
  atomicAddF(d + 1, v.y * sc);
  atomicAddF(d + 2, v.z * sc);
  atomicAddF(d + 3, v.w * sc);
}

// Euler-Maruyama: y += tanh(acc*ndeg + conv_b)*dt + sigma*sqrt(dt)*dW
// 2 elements/thread; also maintains the packed-bf16 mirror of y for the GEMMs.
__global__ void hsde_update(float* __restrict__ y, uint32_t* __restrict__ ybf,
                            const float* __restrict__ acc,
                            const float* __restrict__ node_deg_inv,
                            const float* __restrict__ conv_b,
                            const float* __restrict__ sigma,
                            const int* __restrict__ seed_ptr, int step, int Nn) {
  int tid = blockIdx.x * blockDim.x + threadIdx.x;
  if (tid >= Nn * (LATD / 2)) return;
  int e0 = tid * 2;
  int n = e0 >> 6, c = e0 & 63;
  float nd = node_deg_inv[n];
  uint32_t seed = (uint32_t)seed_ptr[0];
  float2 a = ((const float2*)acc)[tid];
  float2 yv = ((float2*)y)[tid];
  float d0 = tanhf(a.x * nd + conv_b[c]);
  float d1 = tanhf(a.y * nd + conv_b[c + 1]);
  yv.x += d0 * SDE_DT + sigma[c]     * SDE_SQRT_DT * gauss1(seed, (uint32_t)step, (uint32_t)e0);
  yv.y += d1 * SDE_DT + sigma[c + 1] * SDE_SQRT_DT * gauss1(seed, (uint32_t)step, (uint32_t)e0 + 1);
  ((float2*)y)[tid] = yv;
  ybf[tid] = pack2bf(yv.x, yv.y);
}

// ----------------------------- launch --------------------------------------

extern "C" void kernel_launch(void* const* d_in, const int* in_sizes, int n_in,
                              void* d_out, int out_size, void* d_ws, size_t ws_size,
                              hipStream_t stream) {
  const float* x       = (const float*)d_in[0];
  const float* enc_w1  = (const float*)d_in[1];
  const float* enc_b1  = (const float*)d_in[2];
  const float* enc_w2  = (const float*)d_in[3];
  const float* enc_b2  = (const float*)d_in[4];
  const float* conv_w  = (const float*)d_in[5];
  const float* conv_b  = (const float*)d_in[6];
  const float* sigma   = (const float*)d_in[7];
  const float* dec_w1  = (const float*)d_in[8];
  const float* dec_b1  = (const float*)d_in[9];
  const float* dec_w2  = (const float*)d_in[10];
  const float* dec_b2  = (const float*)d_in[11];
  const int*   node_ix = (const int*)d_in[12];
  const int*   edge_ix = (const int*)d_in[13];
  const int*   seed_p  = (const int*)d_in[14];

  const int Nn  = in_sizes[0] / OBSD;
  const int nnz = in_sizes[12];
  const int En  = EDGES;

  // workspace layout (dwords); acc aliases m (m dead after node->edge scatter)
  float* wsf  = (float*)d_ws;
  float*    y    = wsf;
  float*    m    = y    + (size_t)Nn * LATD;      // also "acc"
  float*    ebuf = m    + (size_t)Nn * LATD;
  float*    edeg = ebuf + (size_t)En * LATD;
  float*    ndeg = edeg + En;
  uint32_t* ybf  = (uint32_t*)(ndeg + Nn);        // Nn*32 dwords (packed bf16)
  uint32_t* xbf  = ybf + (size_t)Nn * (LATD / 2);
  uint32_t* wbf  = xbf + (size_t)Nn * (OBSD / 2); // 5 x 2048 dwords
  uint32_t* enc_w1b = wbf;
  uint32_t* enc_w2b = wbf + 2048;
  uint32_t* conv_wb = wbf + 4096;
  uint32_t* dec_w1b = wbf + 6144;
  uint32_t* dec_w2b = wbf + 8192;

  const int tiles  = (Nn + 15) >> 4;
  const int mmBlks = (tiles + WPB - 1) / WPB;
  const int mmThr  = WPB * 32;
  const int scBlks = (nnz * 16 + 255) / 256;
  const int upBlks = (Nn * (LATD / 2) + 255) / 256;
  const int cvBlksN = (Nn * (LATD / 2) + 255) / 256;

  // one-time (per call) conversions: x and the five 64x64 weight matrices
  hsde_cvt_bf16<<<cvBlksN, 256, 0, stream>>>(x, xbf, Nn * (OBSD / 2));
  hsde_cvt_bf16<<<8, 256, 0, stream>>>(enc_w1, enc_w1b, 2048);
  hsde_cvt_bf16<<<8, 256, 0, stream>>>(enc_w2, enc_w2b, 2048);
  hsde_cvt_bf16<<<8, 256, 0, stream>>>(conv_w, conv_wb, 2048);
  hsde_cvt_bf16<<<8, 256, 0, stream>>>(dec_w1, dec_w1b, 2048);
  hsde_cvt_bf16<<<8, 256, 0, stream>>>(dec_w2, dec_w2b, 2048);

  // degree normalization
  hipMemsetAsync(edeg, 0, (size_t)En * sizeof(float), stream);
  hipMemsetAsync(ndeg, 0, (size_t)Nn * sizeof(float), stream);
  hsde_count<<<(nnz + 255) / 256, 256, 0, stream>>>(node_ix, edge_ix, ndeg, edeg, nnz);
  hsde_invert<<<(En + 255) / 256, 256, 0, stream>>>(edeg, En);
  hsde_invert<<<(Nn + 255) / 256, 256, 0, stream>>>(ndeg, Nn);

  // encode, then seed the bf16 mirror of y
  hsde_mlp2<<<mmBlks, mmThr, 0, stream>>>(xbf, enc_w1b, enc_b1, enc_w2b, enc_b2, y, Nn);
  hsde_cvt_bf16<<<cvBlksN, 256, 0, stream>>>(y, ybf, Nn * (LATD / 2));

  // 20 SDE steps
  for (int step = 0; step < SDE_STEPS; ++step) {
    hsde_conv_mm<<<mmBlks, mmThr, 0, stream>>>(ybf, conv_wb, m, Nn);
    hipMemsetAsync(ebuf, 0, (size_t)En * LATD * sizeof(float), stream);
    hsde_scatter<<<scBlks, 256, 0, stream>>>(m, node_ix, edge_ix, nullptr, ebuf, nnz);
    hipMemsetAsync(m, 0, (size_t)Nn * LATD * sizeof(float), stream);   // m becomes acc
    hsde_scatter<<<scBlks, 256, 0, stream>>>(ebuf, edge_ix, node_ix, edeg, m, nnz);
    hsde_update<<<upBlks, 256, 0, stream>>>(y, ybf, m, ndeg, conv_b, sigma, seed_p, step, Nn);
  }

  // decode -> d_out (ybf is in sync with y after the last update)
  hsde_mlp2<<<mmBlks, mmThr, 0, stream>>>(ybf, dec_w1b, dec_b1, dec_w2b, dec_b2,
                                          (float*)d_out, Nn);
}
